// PolyLSTM_66571993088165
// MI455X (gfx1250) — 8-bit matrix operand order repaired and checked by reading, compiled, not run on hardware
//
#include <hip/hip_runtime.h>

typedef __attribute__((ext_vector_type(8))) int   v8i;
typedef __attribute__((ext_vector_type(8))) float v8f;

#define T_STEPS 256
#define HDIM    200
#define RPAD    832
#define NTHR    832
#define NWAVE   26
#define CHL     50
#define KT_N    7
#define KT_REG  5
#define SEQ_ELEMS (T_STEPS * 8 * HDIM)

__device__ __forceinline__ unsigned f32_to_e4m3(float f) {
  union { float f; unsigned u; } v; v.f = f;
  unsigned s = (v.u >> 24) & 0x80u;
  int e = (int)((v.u >> 23) & 0xffu) - 127;
  unsigned m = (v.u >> 20) & 0x7u;
  if (e < -6) return s;
  if (e > 8)  return s | 0x7Eu;
  return s | ((unsigned)(e + 7) << 3) | m;
}

__device__ __forceinline__ float fast_sigmoid(float x) {
  x = fminf(fmaxf(x, -30.f), 30.f);
  return 1.f / (1.f + __expf(-x));
}
__device__ __forceinline__ float fast_tanh(float x) {
  x = fminf(fmaxf(x, -15.f), 15.f);
  float e = __expf(-2.f * x);
  return (1.f - e) / (1.f + e);
}

__global__ void plstm_init_flags(unsigned* __restrict__ prog) {
  int i = threadIdx.x;
  if (i < 200) prog[i] = 0u;
}

__global__ void plstm_mlp1(const float* __restrict__ x, const float* __restrict__ W1,
                           const float* __restrict__ b1, _Float16* __restrict__ seq0) {
  int id = blockIdx.x * 256 + threadIdx.x;
  if (id >= SEQ_ELEMS) return;
  int t = id / 1600; int r = id % 1600; int b = r / 200; int j = r % 200;
  float acc = b1[j];
  const float* xp = x + ((size_t)b * T_STEPS + t) * 6;
  const float* wp = W1 + j * 6;
  #pragma unroll
  for (int k = 0; k < 6; ++k) acc += xp[k] * wp[k];
  seq0[t * 1600 + b * 200 + j] = (_Float16)acc;
}

__global__ __launch_bounds__(NTHR)
void plstm_chunk(const float* __restrict__ Wih, const float* __restrict__ Whh,
                 const float* __restrict__ bih, const float* __restrict__ bhh,
                 const _Float16* __restrict__ inseq,
                 _Float16* __restrict__ lob,
                 unsigned* __restrict__ prog,
                 int baseLayer) {
  const int w    = blockIdx.x;
  const int lg   = baseLayer + w;
  const int tid  = threadIdx.x;
  const int lane = tid & 31;
  const int wv   = tid >> 5;
  const int lg16 = (lane >> 4) & 1;
  const int ln   = lane & 15;

  __shared__ __align__(64) int a_lds[NWAVE][4][32][8];
  __shared__ __align__(64) int bfrag[KT_N][32][8];
  __shared__ float h_lds[1600];
  __shared__ float c_lds[1600];
  __shared__ float gact[RPAD][8];
  __shared__ float bias_s[RPAD];

  const float* wih = Wih + (size_t)lg * (800 * 200);
  const float* whh = Whh + (size_t)lg * (800 * 200);
  v8i afr[2][KT_REG];
  #pragma unroll
  for (int mi = 0; mi < 2; ++mi) {
    int mt   = wv * 2 + mi;
    int pr   = mt * 16 + ln;
    int gate = pr / 208, jj = pr % 208;
    bool valid = (jj < 200);
    int srow = gate * 200 + jj;
    #pragma unroll
    for (int kt = KT_REG; kt < KT_N; ++kt) {
      v8i tmp;
      #pragma unroll
      for (int v = 0; v < 8; ++v) {
        int K0 = kt * 64 + (((v >> 1) & 3) * 16) + lg16 * 8 + ((v & 1) * 4);
        unsigned d = 0u;
        if (valid && K0 < 400) {
          const float* p = (K0 < 200) ? (wih + srow * 200 + K0)
                                      : (whh + srow * 200 + (K0 - 200));
          d  = f32_to_e4m3(p[0]);
          d |= f32_to_e4m3(p[1]) << 8;
          d |= f32_to_e4m3(p[2]) << 16;
          d |= f32_to_e4m3(p[3]) << 24;
        }
        tmp[v] = (int)d;
      }
      *(v8i*)&a_lds[wv][mi * 2 + (kt - KT_REG)][lane][0] = tmp;
    }
    #pragma unroll
    for (int kt = 0; kt < KT_REG; ++kt) {
      #pragma unroll
      for (int v = 0; v < 8; ++v) {
        int K0 = kt * 64 + (((v >> 1) & 3) * 16) + lg16 * 8 + ((v & 1) * 4);
        unsigned d = 0u;
        if (valid) {
          const float* p = (K0 < 200) ? (wih + srow * 200 + K0)
                                      : (whh + srow * 200 + (K0 - 200));
          d  = f32_to_e4m3(p[0]);
          d |= f32_to_e4m3(p[1]) << 8;
          d |= f32_to_e4m3(p[2]) << 16;
          d |= f32_to_e4m3(p[3]) << 24;
        }
        afr[mi][kt][v] = (int)d;
      }
    }
  }
  for (int r = tid; r < RPAD; r += NTHR) {
    int gate = r / 208, jj = r % 208;
    bias_s[r] = (jj < 200) ? (bih[(size_t)lg * 800 + gate * 200 + jj] +
                              bhh[(size_t)lg * 800 + gate * 200 + jj]) : 0.f;
  }
  for (int i = tid; i < 1600; i += NTHR) { h_lds[i] = 0.f; c_lds[i] = 0.f; }
  __syncthreads();

  const _Float16* xsrc = (w == 0) ? inseq : (lob + (size_t)(w - 1) * SEQ_ELEMS);
  _Float16*       lout = lob + (size_t)w * SEQ_ELEMS;

  for (int t = 0; t < T_STEPS; ++t) {
    if (w > 0) {
      if (tid == 0) {
        while (__hip_atomic_load(&prog[lg - 1], __ATOMIC_ACQUIRE,
                                 __HIP_MEMORY_SCOPE_AGENT) < (unsigned)(t + 1)) {
          __builtin_amdgcn_s_sleep(1);
        }
      }
      __syncthreads();
      __threadfence();
    }
    const _Float16* xin = xsrc + (size_t)t * 1600;

    for (int idx = tid; idx < KT_N * 256; idx += NTHR) {
      int kt  = idx >> 8;
      int rem = idx & 255;
      int bl  = rem >> 3;
      int v   = rem & 7;
      int n   = bl & 15;
      int bg  = bl >> 4;
      int K0 = kt * 64 + (((v >> 1) & 3) * 16) + bg * 8 + ((v & 1) * 4);
      unsigned d = 0u;
      if (n < 8 && K0 < 400) {
        float f0, f1, f2, f3;
        if (K0 < 200) {
          const _Float16* p = xin + n * 200 + K0;
          f0 = (float)p[0]; f1 = (float)p[1]; f2 = (float)p[2]; f3 = (float)p[3];
        } else {
          const float* p = h_lds + n * 200 + (K0 - 200);
          f0 = p[0]; f1 = p[1]; f2 = p[2]; f3 = p[3];
        }
        d  = f32_to_e4m3(f0);
        d |= f32_to_e4m3(f1) << 8;
        d |= f32_to_e4m3(f2) << 16;
        d |= f32_to_e4m3(f3) << 24;
      }
      bfrag[kt][bl][v] = (int)d;
    }
    __syncthreads();

    v8f acc0 = {}; v8f acc1 = {};
    #pragma unroll
    for (int kt = 0; kt < KT_REG; ++kt) {
      v8i bv = *((const v8i*)&bfrag[kt][lane][0]);
      acc0 = __builtin_amdgcn_wmma_f32_16x16x64_fp8_fp8(afr[0][kt], bv, (short)0, acc0, false, false);
      acc1 = __builtin_amdgcn_wmma_f32_16x16x64_fp8_fp8(afr[1][kt], bv, (short)0, acc1, false, false);
    }
    #pragma unroll
    for (int kt = KT_REG; kt < KT_N; ++kt) {
      v8i bv = *((const v8i*)&bfrag[kt][lane][0]);
      v8i a0 = *((const v8i*)&a_lds[wv][0 * 2 + (kt - KT_REG)][lane][0]);
      v8i a1 = *((const v8i*)&a_lds[wv][1 * 2 + (kt - KT_REG)][lane][0]);
      acc0 = __builtin_amdgcn_wmma_f32_16x16x64_fp8_fp8(a0, bv, (short)0, acc0, false, false);
      acc1 = __builtin_amdgcn_wmma_f32_16x16x64_fp8_fp8(a1, bv, (short)0, acc1, false, false);
    }

    #pragma unroll
    for (int mi = 0; mi < 2; ++mi) {
      int mt = wv * 2 + mi;
      int gate = mt / 13;
      bool is_g = (gate == 2);
      if (ln < 8) {
        #pragma unroll
        for (int r = 0; r < 8; ++r) {
          int pr = mt * 16 + lg16 * 8 + r;
          float val = (mi ? acc1[r] : acc0[r]) + bias_s[pr];
          gact[pr][ln] = is_g ? fast_tanh(val) : fast_sigmoid(val);
        }
      }
    }
    __syncthreads();

    for (int idx = tid; idx < 1600; idx += NTHR) {
      int n = idx / 200, j = idx % 200;
      float iv = gact[j][n];
      float fv = gact[208 + j][n];
      float gv = gact[416 + j][n];
      float ov = gact[624 + j][n];
      float c  = fv * c_lds[idx] + iv * gv;
      float h  = ov * fast_tanh(c);
      c_lds[idx] = c;
      h_lds[idx] = h;
      lout[(size_t)t * 1600 + idx] = (_Float16)h;
    }
    __threadfence();
    __syncthreads();
    if (tid == 0) {
      __hip_atomic_store(&prog[lg], (unsigned)(t + 1), __ATOMIC_RELEASE,
                         __HIP_MEMORY_SCOPE_AGENT);
    }
  }
}

__global__ void plstm_head(const _Float16* __restrict__ fin,
                           const float* __restrict__ W2a, const float* __restrict__ b2a,
                           const float* __restrict__ W2b, const float* __restrict__ b2b,
                           float* __restrict__ out) {
  __shared__ float hid[50];
  int bid = blockIdx.x;
  int t = bid >> 3, b = bid & 7;
  int tid = threadIdx.x;
  const _Float16* row = fin + (size_t)t * 1600 + b * 200;
  if (tid < 50) {
    float a = b2a[tid];
    const float* wp = W2a + tid * 200;
    for (int j = 0; j < 200; ++j) a += (float)row[j] * wp[j];
    hid[tid] = fmaxf(a, 0.f);
  }
  __syncthreads();
  if (tid < 24) {
    float a = b2b[tid];
    const float* wp = W2b + tid * 50;
    #pragma unroll
    for (int k = 0; k < 50; ++k) a += hid[k] * wp[k];
    out[((size_t)b * T_STEPS + t) * 24 + tid] = a;
  }
}

extern "C" void kernel_launch(void* const* d_in, const int* in_sizes, int n_in,
                              void* d_out, int out_size, void* d_ws, size_t ws_size,
                              hipStream_t stream) {
  (void)in_sizes; (void)n_in; (void)out_size; (void)ws_size;
  const float* x   = (const float*)d_in[0];
  const float* W1  = (const float*)d_in[1];
  const float* b1  = (const float*)d_in[2];
  const float* Wih = (const float*)d_in[3];
  const float* Whh = (const float*)d_in[4];
  const float* bih = (const float*)d_in[5];
  const float* bhh = (const float*)d_in[6];
  const float* W2a = (const float*)d_in[7];
  const float* b2a = (const float*)d_in[8];
  const float* W2b = (const float*)d_in[9];
  const float* b2b = (const float*)d_in[10];

  unsigned*  prog = (unsigned*)d_ws;
  _Float16*  seq0 = (_Float16*)((char*)d_ws + 1024);
  _Float16*  lob  = seq0 + SEQ_ELEMS;

  plstm_init_flags<<<1, 256, 0, stream>>>(prog);
  plstm_mlp1<<<(SEQ_ELEMS + 255) / 256, 256, 0, stream>>>(x, W1, b1, seq0);

  for (int c = 0; c < 200 / CHL; ++c) {
    const _Float16* in = (c == 0) ? seq0 : (lob + (size_t)(CHL - 1) * SEQ_ELEMS);
    plstm_chunk<<<CHL, NTHR, 0, stream>>>(Wih, Whh, bih, bhh, in, lob, prog, c * CHL);
  }

  plstm_head<<<T_STEPS * 8, 64, 0, stream>>>(lob + (size_t)(CHL - 1) * SEQ_ELEMS,
                                             W2a, b2a, W2b, b2b, (float*)d_out);
}
